// ContrastLoss_1898375545407
// MI455X (gfx1250) — compile-verified, matrix operand order checked by reading, not run on hardware
//
#include <hip/hip_runtime.h>
#include <math.h>
#include <stdint.h>

// ---------------------------------------------------------------------------
// ContrastLoss on MI455X (gfx1250).
// Band-limited DFT as a bf16-split WMMA GEMM; Gram-collapsed comparisons.
// B-tile staging uses CDNA5 async global->LDS loads (ASYNCcnt path).
// Workspace usage: ~63.1 MB.
// ---------------------------------------------------------------------------

typedef __attribute__((ext_vector_type(16))) __bf16 v16bf;
typedef __attribute__((ext_vector_type(8)))  __bf16 v8bf;
typedef __attribute__((ext_vector_type(8)))  float  v8f;

#define T_LEN   32768
#define L_CROP  4096
#define NF      477      // band bins [92, 568]
#define F0      92
#define NBINS   954      // cos + sin columns
#define NPAD    960      // padded to multiple of 64
#define NCROPS  8192     // 4096 ST + 4096 GT
#define NSET    2048     // crops per set (C*K)

// ------------------------------- basis -------------------------------------
__global__ __launch_bounds__(256) void basis_kernel(__bf16* __restrict__ hi,
                                                    __bf16* __restrict__ lo) {
  size_t idx = (size_t)blockIdx.x * 256 + threadIdx.x;   // NPAD * L_CROP
  if (idx >= (size_t)NPAD * L_CROP) return;
  int j = (int)(idx >> 12);
  int t = (int)(idx & 4095);
  float v = 0.0f;
  if (j < NBINS) {
    int f  = (j < NF) ? j : (j - NF);
    int kf = F0 + f;
    int r  = (kf * t) & 4095;                 // exact integer angle reduction
    float theta = (float)r * (float)(6.283185307179586 / 4096.0);
    v = (j < NF) ? cosf(theta) : sinf(theta);
  }
  __bf16 h = (__bf16)v;
  hi[idx] = h;
  lo[idx] = (__bf16)(v - (float)h);
}

// ------------------------------ row pointers --------------------------------
__global__ __launch_bounds__(256) void rowptr_kernel(
    const float* __restrict__ mo, const float* __restrict__ gt,
    const int* __restrict__ offst, const int* __restrict__ offt,
    unsigned long long* __restrict__ rowptr) {
  int m = blockIdx.x * 256 + threadIdx.x;
  if (m >= NCROPS) return;
  const float* p;
  if (m < 4096) {
    int b = m >> 11, c = (m >> 5) & 63, k = m & 31;
    int off = offst[((b * 64) + c) * 32 + k];
    p = mo + (size_t)(b * 64 + c) * T_LEN + off;
  } else {
    int mm = m - 4096;
    int b = mm >> 11, c = (mm >> 5) & 63, k = mm & 31;
    int off = offt[((b * 64) + c) * 32 + k];
    p = gt + (size_t)b * T_LEN + off;
  }
  rowptr[m] = (unsigned long long)(uintptr_t)p;
}

// ------------------------------- DFT GEMM -----------------------------------
// C[8192 x 960] = A[8192 x 4096] * B[4096 x 960], bf16 hi/lo split, f32 acc.
// Block tile: M=128 (8 waves x 16), N=64 (4 x 16), K-step 32.
__global__ __launch_bounds__(256) void dft_gemm_kernel(
    const unsigned long long* __restrict__ rowptr,
    const __bf16* __restrict__ basisHi,
    const __bf16* __restrict__ basisLo,
    float* __restrict__ X) {
  __shared__ __bf16 sAhi[128 * 32];
  __shared__ __bf16 sAlo[128 * 32];
  __shared__ __bf16 sBhi[64 * 32];
  __shared__ __bf16 sBlo[64 * 32];

  const int tid   = threadIdx.x;
  const int ntile = blockIdx.x;   // 0..14
  const int mtile = blockIdx.y;   // 0..63
  const int wave  = tid >> 5;
  const int lane  = tid & 31;
  const int half  = lane >> 4;    // 0/1
  const int lr    = lane & 15;
  const int c0    = half * 8;     // k-chunk base per ISA 16-bit frag layout

  // A staging: 2 threads per row, 16 consecutive floats each (offset may be
  // only 4B-aligned -> scalar dword loads).
  const int ar = tid >> 1;
  const int ah = tid & 1;
  const float* aptr =
      (const float*)(uintptr_t)rowptr[mtile * 128 + ar] + ah * 16;

  // B staging: async global->LDS, one 16B chunk per thread per array
  // (256 threads x 16B = full 64x32 bf16 tile).
  const int bn = tid >> 2;
  const int bq = tid & 3;
  const size_t bgl = (size_t)(ntile * 64 + bn) * L_CROP + bq * 8;
  // Low 32 bits of a flat shared pointer are the LDS byte address
  // (ISA 10.2: LDS aperture truncates addr[31:0]).
  const unsigned ldsBhi = (unsigned)(uintptr_t)&sBhi[bn * 32 + bq * 8];
  const unsigned ldsBlo = (unsigned)(uintptr_t)&sBlo[bn * 32 + bq * 8];

  v8f acc[4] = {v8f{}, v8f{}, v8f{}, v8f{}};

  for (int k0 = 0; k0 < L_CROP; k0 += 32) {
    // ---- issue async B-tile copy (ASYNCcnt) ----
    {
      const __bf16* gh = basisHi + bgl + k0;
      const __bf16* gl = basisLo + bgl + k0;
      asm volatile(
          "global_load_async_to_lds_b128 %0, %2, off\n\t"
          "global_load_async_to_lds_b128 %1, %3, off"
          :
          : "v"(ldsBhi), "v"(ldsBlo), "v"(gh), "v"(gl)
          : "memory");
    }
    // ---- stage A (gather + f32 -> bf16 hi/lo split) while B copies ----
    {
      const float* p = aptr + k0;
      __bf16* dh = &sAhi[ar * 32 + ah * 16];
      __bf16* dl = &sAlo[ar * 32 + ah * 16];
#pragma unroll
      for (int i = 0; i < 16; ++i) {
        float v  = p[i];
        __bf16 h = (__bf16)v;
        dh[i] = h;
        dl[i] = (__bf16)(v - (float)h);
      }
    }
    asm volatile("s_wait_asynccnt 0x0" ::: "memory");
    __syncthreads();

    // A fragments: lane holds row (lane&15); element i <-> k = (i<8?i:i+8)+c0.
    const int arow = wave * 16 + lr;
    v8bf a0 = *(const v8bf*)&sAhi[arow * 32 + c0];
    v8bf a1 = *(const v8bf*)&sAhi[arow * 32 + c0 + 16];
    v16bf afh = __builtin_shufflevector(a0, a1, 0, 1, 2, 3, 4, 5, 6, 7, 8, 9,
                                        10, 11, 12, 13, 14, 15);
    v8bf a2 = *(const v8bf*)&sAlo[arow * 32 + c0];
    v8bf a3 = *(const v8bf*)&sAlo[arow * 32 + c0 + 16];
    v16bf afl = __builtin_shufflevector(a2, a3, 0, 1, 2, 3, 4, 5, 6, 7, 8, 9,
                                        10, 11, 12, 13, 14, 15);

#pragma unroll
    for (int nb = 0; nb < 4; ++nb) {
      const int brow = nb * 16 + lr;   // B stored transposed: [n][k]
      v8bf b0 = *(const v8bf*)&sBhi[brow * 32 + c0];
      v8bf b1 = *(const v8bf*)&sBhi[brow * 32 + c0 + 16];
      v16bf bfh = __builtin_shufflevector(b0, b1, 0, 1, 2, 3, 4, 5, 6, 7, 8, 9,
                                          10, 11, 12, 13, 14, 15);
      v8bf b2 = *(const v8bf*)&sBlo[brow * 32 + c0];
      v8bf b3 = *(const v8bf*)&sBlo[brow * 32 + c0 + 16];
      v16bf bfl = __builtin_shufflevector(b2, b3, 0, 1, 2, 3, 4, 5, 6, 7, 8, 9,
                                          10, 11, 12, 13, 14, 15);
      acc[nb] = __builtin_amdgcn_wmma_f32_16x16x32_bf16(
          false, afh, false, bfh, (short)0, acc[nb], false, false);
      acc[nb] = __builtin_amdgcn_wmma_f32_16x16x32_bf16(
          false, afh, false, bfl, (short)0, acc[nb], false, false);
      acc[nb] = __builtin_amdgcn_wmma_f32_16x16x32_bf16(
          false, afl, false, bfh, (short)0, acc[nb], false, false);
    }
    __syncthreads();
  }

  // D layout: lane<16 -> M=r, N=lane ; lane>=16 -> M=8+r, N=lane-16.
  const size_t mrow = (size_t)mtile * 128 + wave * 16 + half * 8;
#pragma unroll
  for (int nb = 0; nb < 4; ++nb) {
    float* xp = X + mrow * NPAD + (size_t)(ntile * 64 + nb * 16 + lr);
#pragma unroll
    for (int r = 0; r < 8; ++r) xp[(size_t)r * NPAD] = acc[nb][r];
  }
}

// ---------------------------- reduction helper ------------------------------
__device__ inline float block_reduce_sum(float v, float* red) {
  int tid = threadIdx.x;
  red[tid] = v;
  __syncthreads();
  for (int s = 128; s > 0; s >>= 1) {
    if (tid < s) red[tid] += red[tid + s];
    __syncthreads();
  }
  float r = red[0];
  __syncthreads();
  return r;
}

// ------------------------------- PSD ----------------------------------------
// q[f] = cos^2 + sin^2 ; s = q / sum(q). (1/L norm and mean-sub cancel/vanish)
__global__ __launch_bounds__(256) void psd_kernel(const float* __restrict__ X,
                                                  float* __restrict__ S,
                                                  float* __restrict__ ssArr) {
  __shared__ float q[480];
  __shared__ float red[256];
  const int m = blockIdx.x;
  const int tid = threadIdx.x;
  const float* xr = X + (size_t)m * NPAD;
  for (int f = tid; f < 480; f += 256) {
    if (f < NF) {
      float c = xr[f];
      float s = xr[NF + f];
      q[f] = c * c + s * s;
    } else {
      q[f] = 0.0f;
    }
  }
  __syncthreads();
  float loc = 0.0f;
  for (int f = tid; f < 480; f += 256) loc += q[f];
  float total = block_reduce_sum(loc, red);
  float inv = 1.0f / total;
  float ss = 0.0f;
  for (int f = tid; f < 480; f += 256) {
    float sv = q[f] * inv;
    S[(size_t)m * 480 + f] = sv;
    ss += sv * sv;
  }
  float sst = block_reduce_sum(ss, red);
  if (tid == 0) ssArr[m] = sst;
}

// ----------------------- deterministic column sums --------------------------
__global__ __launch_bounds__(256) void reducev_kernel(const float* __restrict__ S,
                                                      float* __restrict__ V) {
  int idx = blockIdx.x * 256 + threadIdx.x;   // 4 sets * 480 features
  if (idx >= 4 * 480) return;
  int set = idx / 480, f = idx % 480;
  const float* base = S + (size_t)set * NSET * 480 + f;
  float acc = 0.0f;
  for (int i = 0; i < NSET; ++i) acc += base[(size_t)i * 480];
  V[idx] = acc;
}

// ------------------------------- finalize -----------------------------------
__global__ __launch_bounds__(256) void finalize_kernel(
    const float* __restrict__ V, const float* __restrict__ ssArr,
    const float* __restrict__ label_flag, float* __restrict__ out) {
  __shared__ float red[256];
  const int tid = threadIdx.x;
  // dots needed: (0,0)(1,1)(0,1)(0,2)(1,3)(1,2)(0,3)  [0=s0,1=s1,2=g0,3=g1]
  const int pa[7] = {0, 1, 0, 0, 1, 1, 0};
  const int pb[7] = {0, 1, 1, 2, 3, 2, 3};
  float dots[7];
  for (int p = 0; p < 7; ++p) {
    float loc = 0.0f;
    for (int f = tid; f < 480; f += 256)
      loc += V[pa[p] * 480 + f] * V[pb[p] * 480 + f];
    dots[p] = block_reduce_sum(loc, red);
  }
  float ss[4];
  for (int s = 0; s < 4; ++s) {
    float loc = 0.0f;
    for (int i = tid; i < NSET; i += 256) loc += ssArr[s * NSET + i];
    ss[s] = block_reduce_sum(loc, red);
  }
  if (tid == 0) {
    const double N = (double)NSET, F = (double)NF;
    auto cmpAA = [&](double s2, double d) {
      return (2.0 * N * s2 - 2.0 * d) / F / (N * (N - 1.0));
    };
    auto cmpAB = [&](double sa, double sb, double d) {
      return (N * sa + N * sb - 2.0 * d) / (F * N * N);
    };
    double pos_loss = 0.5 * (cmpAA(ss[0], dots[0]) + cmpAA(ss[1], dots[1]));
    double neg_loss = -cmpAB(ss[0], ss[1], dots[2]);
    float lf0 = label_flag[0], lf1 = label_flag[1];
    float lsum = lf0 + lf1;
    double denom = (lsum == 0.0f) ? 1.0 : (double)lsum;
    double pos_GT = ((double)lf0 * cmpAB(ss[0], ss[2], dots[3]) +
                     (double)lf1 * cmpAB(ss[1], ss[3], dots[4])) / denom;
    double neg_GT = -((double)lf0 * cmpAB(ss[1], ss[2], dots[5]) +
                      (double)lf1 * cmpAB(ss[0], ss[3], dots[6])) / denom;
    if (lsum == 0.0f) { pos_GT = 0.0; neg_GT = 0.0; }
    out[0] = (float)(pos_loss + neg_loss + pos_GT + neg_GT);
    out[1] = (float)pos_loss;
    out[2] = (float)neg_loss;
    out[3] = (float)pos_GT;
    out[4] = (float)neg_GT;
  }
}

// ------------------------------- launch -------------------------------------
extern "C" void kernel_launch(void* const* d_in, const int* in_sizes, int n_in,
                              void* d_out, int out_size, void* d_ws,
                              size_t ws_size, hipStream_t stream) {
  const float* model_output = (const float*)d_in[0];
  const float* GT_sig       = (const float*)d_in[1];
  const float* label_flag   = (const float*)d_in[2];
  const int*   offsets_st   = (const int*)d_in[3];
  const int*   offsets_t    = (const int*)d_in[4];
  float* out = (float*)d_out;

  // Workspace layout (bytes), 256-aligned sections, total ~63.1 MB.
  char* ws = (char*)d_ws;
  __bf16* basisHi = (__bf16*)(ws + 0);                        //  7,864,320
  __bf16* basisLo = (__bf16*)(ws + 7864320);                  //  7,864,320
  unsigned long long* rowptr = (unsigned long long*)(ws + 15728640); // 65,536
  float* X     = (float*)(ws + 15794176);                     // 31,457,280
  float* S     = (float*)(ws + 47251456);                     // 15,728,640
  float* ssArr = (float*)(ws + 62980096);                     //     32,768
  float* V     = (float*)(ws + 63012864);                     //      7,680

  basis_kernel<<<(NPAD * L_CROP) / 256, 256, 0, stream>>>(basisHi, basisLo);
  rowptr_kernel<<<NCROPS / 256, 256, 0, stream>>>(model_output, GT_sig,
                                                  offsets_st, offsets_t, rowptr);
  dft_gemm_kernel<<<dim3(15, 64), 256, 0, stream>>>(rowptr, basisHi, basisLo, X);
  psd_kernel<<<NCROPS, 256, 0, stream>>>(X, S, ssArr);
  reducev_kernel<<<8, 256, 0, stream>>>(S, V);
  finalize_kernel<<<1, 256, 0, stream>>>(V, ssArr, label_flag, out);
}
